// DownModule_2972117369413
// MI455X (gfx1250) — matrix operand order repaired and checked by reading, not compiled, not run on hardware
//
#include <hip/hip_runtime.h>

typedef __attribute__((ext_vector_type(16))) _Float16 v16h;
typedef __attribute__((ext_vector_type(8)))  float    v8f;
typedef __attribute__((ext_vector_type(4)))  float    v4f;

#define BN_EPS 1e-5f

// Fused gather -> pointwise conv (WMMA f16, f32 accum) -> BN -> ReLU -> max over K.
// One wave (32 lanes) processes one voxel per loop iteration:
//   A (16x32 f16) = gathered neighbor features (2 fragments: neighbors 0-15, 16-31)
//   B (32x16 f16) = W^T tile, 4 fragments covering C_out = 64 (loop-invariant)
//   D (16x16 f32) accum, per-lane all 8 elements share one output channel.
//
// MI455X-specific choices:
//  - 51.2 MB feature table is L2-resident (192 MB L2): gathers use default
//    (temporal) policy so they hit L2.
//  - indices / masks / output are single-touch streams: non-temporal hints so
//    they do not evict the feature table from L2.
//  - ReLU commutes with max, applied once after the neighbor reduction.
__global__ __launch_bounds__(256) void down_fused_wmma(
    const float* __restrict__ vf,      // [N, 32] voxel features
    const int*   __restrict__ kidx,    // [M, 32] neighbor indices
    const int*   __restrict__ kmask,   // [M, 32] 1 = invalid
    const float* __restrict__ W,       // [64, 32]
    const float* __restrict__ bias,    // [64]
    const float* __restrict__ gamma,   // [64]
    const float* __restrict__ beta,    // [64]
    const float* __restrict__ mean,    // [64]
    const float* __restrict__ var,     // [64]
    float*       __restrict__ out,     // [M, 64]
    int M, int total_waves)
{
    const int lane = threadIdx.x & 31;
    const int wave = (int)((blockIdx.x * blockDim.x + threadIdx.x) >> 5);
    const int col  = lane & 15;   // B/D column within 16-wide tile
    const int g    = lane >> 4;   // lane half selects K sub-range

    // ---- Loop-invariant: B fragments (W^T tiles) + fused BN affine ----
    // B 16-bit 32x16 layout: lane holds column `col`; half h<8 -> K = g*8 + h; h>=8 -> K = 16 + g*8 + (h-8).
    v16h bfrag[4];
    float sc[4], sh[4];
    #pragma unroll
    for (int q = 0; q < 4; ++q) {
        const int o = q * 16 + col;               // output channel
        const float* wp = W + o * 32 + g * 8;     // W[o][g*8 .. g*8+7], W[o][16+g*8 .. 16+g*8+7]
        v16h bq;
        #pragma unroll
        for (int h = 0; h < 16; ++h) bq[h] = (_Float16)wp[h < 8 ? h : h + 8];
        bfrag[q] = bq;
        const float s = gamma[o] * rsqrtf(var[o] + BN_EPS);
        sc[q] = s;
        sh[q] = (bias[o] - mean[o]) * s + beta[o];  // folded bias + BN shift
    }

    for (int m = wave; m < M; m += total_waves) {
        // Prefetch next iteration's index row (speculative; dropped if OOB).
        __builtin_prefetch(kidx + (size_t)(m + total_waves) * 32 + lane, 0, 0);

        // ---- Gather A fragments ----
        // A 16-bit 16x32 layout: lane holds neighbor row (col + 16*f);
        // half h<8 -> channel g*8+h ; h>=8 -> channel 16+g*8+(h-8).
        v16h a[2];
        #pragma unroll
        for (int f = 0; f < 2; ++f) {
            const int r  = col + 16 * f;                    // neighbor index in [0,32)
            const int ki = __builtin_nontemporal_load(kidx  + (size_t)m * 32 + r);
            const int km = __builtin_nontemporal_load(kmask + (size_t)m * 32 + r);
            const float msk = (km == 0) ? 1.0f : 0.0f;
            const float* rp = vf + (size_t)(ki < 0 ? 0 : ki) * 32 + g * 8;
            const v4f lo0 = *(const v4f*)(rp +  0);
            const v4f lo1 = *(const v4f*)(rp +  4);
            const v4f hi0 = *(const v4f*)(rp + 16);
            const v4f hi1 = *(const v4f*)(rp + 20);
            v16h af;
            #pragma unroll
            for (int j = 0; j < 4; ++j) {
                af[j     ] = (_Float16)(lo0[j] * msk);
                af[j +  4] = (_Float16)(lo1[j] * msk);
                af[j +  8] = (_Float16)(hi0[j] * msk);
                af[j + 12] = (_Float16)(hi1[j] * msk);
            }
            a[f] = af;
        }

        // ---- 8x WMMA + fused epilogue ----
        #pragma unroll
        for (int q = 0; q < 4; ++q) {
            v8f cz = {};
            v8f d0 = __builtin_amdgcn_wmma_f32_16x16x32_f16(
                false, a[0], false, bfrag[q], (short)0, cz, false, false);
            v8f d1 = __builtin_amdgcn_wmma_f32_16x16x32_f16(
                false, a[1], false, bfrag[q], (short)0, cz, false, false);

            // Per-lane channel o = q*16+col; BN affine + max over the 8
            // neighbor rows held by this lane in each fragment. ReLU is
            // monotone, so it is applied once after the reduction.
            float pm = -3.402823466e+38f;
            #pragma unroll
            for (int e = 0; e < 8; ++e) {
                pm = fmaxf(pm, d0[e] * sc[q] + sh[q]);
                pm = fmaxf(pm, d1[e] * sc[q] + sh[q]);
            }
            // Combine lane L with L+16 (same channel, other neighbor block).
            pm = fmaxf(pm, __shfl_xor(pm, 16, 32));
            pm = fmaxf(pm, 0.0f);  // ReLU
            if (lane < 16)
                __builtin_nontemporal_store(pm, out + (size_t)m * 64 + q * 16 + lane);
        }
    }
}

extern "C" void kernel_launch(void* const* d_in, const int* in_sizes, int n_in,
                              void* d_out, int out_size, void* d_ws, size_t ws_size,
                              hipStream_t stream) {
    const float* vf    = (const float*)d_in[0];  // voxel_features [N,32]
    const int*   kidx  = (const int*)  d_in[1];  // key_indices   [M,32]
    const int*   kmask = (const int*)  d_in[2];  // key_mask      [M,32]
    const float* W     = (const float*)d_in[3];  // [64,32]
    const float* bias  = (const float*)d_in[4];  // [64]
    const float* gamma = (const float*)d_in[5];
    const float* beta  = (const float*)d_in[6];
    const float* mean  = (const float*)d_in[7];
    const float* var   = (const float*)d_in[8];
    float* out = (float*)d_out;

    const int M = in_sizes[1] / 32;              // key_indices is [M, K=32]

    int blocks = (M + 7) / 8;                    // 8 waves per 256-thread block
    if (blocks > 2048) blocks = 2048;            // persistent-ish: amortize B frags
    const int total_waves = blocks * 8;

    down_fused_wmma<<<blocks, 256, 0, stream>>>(
        vf, kidx, kmask, W, bias, gamma, beta, mean, var, out, M, total_waves);
}
